// Vrnn_48266842472920
// MI455X (gfx1250) — matrix operand order repaired and checked by reading, not compiled, not run on hardware
//
#include <hip/hip_runtime.h>
#include <hip/hip_bf16.h>

// ---------------------------------------------------------------------------
// Types for CDNA5 WMMA (wave32): D(16x16 f32) = A(16x32 bf16) * B(32x16 bf16) + C
// ---------------------------------------------------------------------------
typedef __attribute__((ext_vector_type(8)))  __bf16 v8bf;
typedef __attribute__((ext_vector_type(16))) __bf16 v16bf;
typedef __attribute__((ext_vector_type(8)))  float  v8f;

__device__ __forceinline__ v8f wmma_bf16(v16bf a, v16bf b, v8f c) {
  // 8 args: (neg_a, A, neg_b, B, c_mod, C, reuse_a, reuse_b)
  return __builtin_amdgcn_wmma_f32_16x16x32_bf16(false, a, false, b, (short)0, c,
                                                 false, false);
}

// Async global->LDS copy, 16 bytes per issue (GLOBAL_LOAD_ASYNC_TO_LDS_B128,
// tracked by ASYNCcnt). LDS address = low 32 bits of the generic pointer
// (flat aperture maps addr[31:0] to the wave's LDS).
__device__ __forceinline__ void async_b128(void* lds, const void* gptr) {
  unsigned l = (unsigned)(unsigned long long)lds;
  asm volatile("global_load_async_to_lds_b128 %0, %1, off"
               :: "v"(l), "v"(gptr) : "memory");
}
__device__ __forceinline__ void async_wait0() {
  asm volatile("s_wait_asynccnt 0x0" ::: "memory");
}

__device__ __forceinline__ float sigf(float x) { return 1.0f / (1.0f + __expf(-x)); }

__device__ __forceinline__ float actf(float x, int act) {
  if (act == 1) return x > 0.0f ? x : 0.0f;                  // relu
  if (act == 2) return (x > 20.0f) ? x : log1pf(__expf(x));  // softplus
  return x;
}

// ---------------------------------------------------------------------------
// Weight prepack: row-major fp32 W[K][N] -> bf16 fragments.
// Fragment layout (ISA 7.12.2, 16-bit B 32x16): for tile (nt,kt), lane l holds
// column n = nt*16 + (l&15), K-values kt*32 + (l>>4)*8 + (e&7) + (e>>3)*16, e=0..15, stored
// contiguously so a B-fragment is one 32-byte per-lane load (coalesced 1KB/wave).
// ---------------------------------------------------------------------------
__global__ void pack_w_kernel(const float* __restrict__ W, __bf16* __restrict__ out,
                              int K, int N) {
  int slot = blockIdx.x * blockDim.x + threadIdx.x;
  int nslots = (K * N) >> 4;
  if (slot >= nslots) return;
  int lane = slot & 31;
  int t = slot >> 5;
  int KT = K >> 5;
  int kt = t % KT;
  int nt = t / KT;
  int k0 = kt * 32 + ((lane >> 4) * 8);
  int n  = nt * 16 + (lane & 15);
  size_t o = (size_t)slot * 16;
#pragma unroll
  for (int e = 0; e < 16; ++e)
    out[o + e] = (__bf16)W[(size_t)(k0 + (e & 7) + ((e >> 3) * 16)) * N + n];
}

__global__ void f2bf_kernel(const float* __restrict__ in, __bf16* __restrict__ out,
                            size_t n) {
  size_t i = (size_t)blockIdx.x * blockDim.x + threadIdx.x;
  if (i < n) out[i] = (__bf16)in[i];
}

// z = mu + softplus_sig * eps   (sig already softplus'd by GEMM epilogue)
__global__ void z_kernel(const float* __restrict__ mu, const float* __restrict__ sig,
                         const float* __restrict__ eps, __bf16* __restrict__ z,
                         size_t n) {
  size_t i = (size_t)blockIdx.x * blockDim.x + threadIdx.x;
  if (i < n) z[i] = (__bf16)(mu[i] + sig[i] * eps[i]);
}

// ---------------------------------------------------------------------------
// Generic WMMA GEMM: C[M,N] = act(A[M,K] @ W[K,N] + bias)
// Block: 256 threads = 8 waves; tile 64(M) x 128(N); waves = 4(M) x 2(N),
// each wave owns 16 rows x 64 cols = 4 accumulators, reusing its A fragment.
// A tiles double-buffered in LDS via async-to-LDS: issue k+1, compute k;
// one barrier + one asynccnt wait per K-step, fetch overlapped with WMMA.
// ---------------------------------------------------------------------------
__global__ __launch_bounds__(256) void gemm_bf16_kernel(
    const __bf16* __restrict__ A, int lda,
    const __bf16* __restrict__ Wp, int KT,
    const float* __restrict__ bias,
    float* __restrict__ Cf, int ldcf,
    __bf16* __restrict__ Cb, int ldcb, int cboff,
    __bf16* __restrict__ Cb2, int ldcb2, int cboff2,
    int act) {
  __shared__ __bf16 At[2][64 * 40];  // stride 40 (80B) -> conflict-free b128 reads

  const int tid = threadIdx.x;
  const int wave = tid >> 5, lane = tid & 31;
  const int mrow = wave & 3, ncol = wave >> 2;
  const size_t m0 = (size_t)blockIdx.y * 64;
  const int ntile0 = blockIdx.x * 8 + ncol * 4;

  const int arow = tid >> 2;       // 0..63
  const int acol = (tid & 3) * 8;  // 0,8,16,24

  const __bf16* gabase = A + (m0 + arow) * (size_t)lda + acol;
  __bf16* ldsdst0 = &At[0][arow * 40 + acol];
  __bf16* ldsdst1 = &At[1][arow * 40 + acol];

  // prologue: stage tile 0
  async_b128(ldsdst0, gabase);

  v8f acc[4] = {};
  const int m  = mrow * 16 + (lane & 15);
  const int kb = (lane >> 4) * 8;

  for (int kt = 0; kt < KT; ++kt) {
    async_wait0();
    __syncthreads();  // tile kt visible workgroup-wide; buffers free to reuse
    if (kt + 1 < KT)
      async_b128((kt + 1) & 1 ? ldsdst1 : ldsdst0, gabase + (kt + 1) * 32);

    const __bf16* buf = At[kt & 1];
    // A fragment per ISA 7.12.2 (16-bit A 16x32): lane m=lane&15,
    // K = kb..kb+7 and kb+16..kb+23, kb = (lane>>4)*8
    v8bf alo = *(const v8bf*)(&buf[m * 40 + kb]);
    v8bf ahi = *(const v8bf*)(&buf[m * 40 + kb + 16]);
    v16bf afrag = __builtin_shufflevector(alo, ahi, 0, 1, 2, 3, 4, 5, 6, 7, 8, 9,
                                          10, 11, 12, 13, 14, 15);
#pragma unroll
    for (int j = 0; j < 4; ++j) {
      const __bf16* gb = Wp + ((size_t)(ntile0 + j) * KT + kt) * 512 + lane * 16;
      v16bf bfrag = *(const v16bf*)gb;  // 32B per lane, coalesced
      acc[j] = wmma_bf16(afrag, bfrag, acc[j]);
    }
  }

  // Epilogue. D layout: lane l, acc elem v -> m=(l>>4)*8+v, n=(l&15)
  const int n16 = lane & 15;
  const int mbase = mrow * 16 + ((lane >> 4) * 8);
#pragma unroll
  for (int j = 0; j < 4; ++j) {
    const int n = (ntile0 + j) * 16 + n16;
    const float bb = bias ? bias[n] : 0.0f;
#pragma unroll
    for (int v = 0; v < 8; ++v) {
      const size_t mm = m0 + mbase + v;
      float val = actf(acc[j][v] + bb, act);
      if (Cf)  Cf[mm * (size_t)ldcf + n] = val;
      if (Cb)  Cb[mm * (size_t)ldcb + cboff + n] = (__bf16)val;
      if (Cb2) Cb2[mm * (size_t)ldcb2 + cboff2 + n] = (__bf16)val;
    }
  }
}

// ---------------------------------------------------------------------------
// Small GRU recurrence (units=128, Keras reset_after=True).
// Grid: 4 blocks, each owns a 16-row batch slice (batch rows independent).
// Entire packed U (96KB) lives in LDS (CDNA5: 320KB/WGP). 8 waves; wave w
// owns gate columns [16w,16w+16) of z,r,hh (tiles w, w+8, w+16).
// xp tiles double-buffered in LDS, prefetched async while step t computes.
// ---------------------------------------------------------------------------
__global__ __launch_bounds__(256) void gru128_kernel(
    const float* __restrict__ xp,    // [B*T, 384], row = b*T + t
    const __bf16* __restrict__ Upk,  // packed [24][4][32][16]
    const float* __restrict__ b1,    // [384]
    __bf16* __restrict__ hs,         // out bf16, row stride ldh, col offset
    int ldh, int coloff, int T) {
  extern __shared__ char smem[];
  __bf16* Ul   = (__bf16*)smem;                     // 49152 el = 98304B
  float*  hF   = (float*)(smem + 98304);            // 16x128 fp32 (8192B)
  __bf16* hB   = (__bf16*)(smem + 106496);          // 16x136 bf16 (4352B)
  float*  xpl0 = (float*)(smem + 110848);           // 16x384 fp32 (24576B)
  float*  xpl1 = (float*)(smem + 135424);           // 16x384 fp32 (24576B)

  const int tid = threadIdx.x, wave = tid >> 5, lane = tid & 31;
  const int b0 = blockIdx.x * 16;

  // prologue: async U -> LDS, zero h, async xp(0)
  for (int i = tid; i < 6144; i += 256)  // 49152/8
    async_b128(Ul + (size_t)i * 8, Upk + (size_t)i * 8);
  for (int i = tid; i < 16 * 128; i += 256) hF[i] = 0.0f;
  for (int i = tid; i < 16 * 136; i += 256) hB[i] = (__bf16)0.0f;
  for (int i = tid; i < 1536; i += 256) {
    int r = (i * 4) / 384, c = (i * 4) % 384;
    async_b128(xpl0 + r * 384 + c, xp + ((size_t)(b0 + r) * T + 0) * 384 + c);
  }
  async_wait0();
  __syncthreads();

  const int n16 = lane & 15;
  const int n = wave * 16 + n16;  // column in [0,128)
  const int mb = (lane >> 4) * 8;
  const float bz_ = b1[n], br_ = b1[128 + n], bh_ = b1[256 + n];

  for (int t = 0; t < T; ++t) {
    // prefetch xp(t+1) into the other buffer (overlaps with this step's WMMAs)
    if (t + 1 < T) {
      float* nxt = ((t + 1) & 1) ? xpl1 : xpl0;
      for (int i = tid; i < 1536; i += 256) {
        int r = (i * 4) / 384, c = (i * 4) % 384;
        async_b128(nxt + r * 384 + c,
                   xp + ((size_t)(b0 + r) * T + (t + 1)) * 384 + c);
      }
    }

    v8f az = {}, ar = {}, ah = {};
#pragma unroll
    for (int kt = 0; kt < 4; ++kt) {
      const int m = lane & 15;
      const int base = kt * 32 + ((lane >> 4) * 8);
      v8bf lo = *(const v8bf*)(hB + m * 136 + base);
      v8bf hi = *(const v8bf*)(hB + m * 136 + base + 16);
      v16bf a = __builtin_shufflevector(lo, hi, 0, 1, 2, 3, 4, 5, 6, 7, 8, 9, 10,
                                        11, 12, 13, 14, 15);
      v16bf bz = *(const v16bf*)(Ul + ((size_t)(wave)      * 4 + kt) * 512 + lane * 16);
      v16bf br = *(const v16bf*)(Ul + ((size_t)(wave + 8)  * 4 + kt) * 512 + lane * 16);
      v16bf bh = *(const v16bf*)(Ul + ((size_t)(wave + 16) * 4 + kt) * 512 + lane * 16);
      az = wmma_bf16(a, bz, az);
      ar = wmma_bf16(a, br, ar);
      ah = wmma_bf16(a, bh, ah);
    }
    __syncthreads();  // all waves done reading hB before updates

    const float* cur = (t & 1) ? xpl1 : xpl0;
#pragma unroll
    for (int v = 0; v < 8; ++v) {
      const int m = mb + v;
      float xz = cur[m * 384 + n];
      float xr = cur[m * 384 + 128 + n];
      float xh = cur[m * 384 + 256 + n];
      float z = sigf(xz + az[v] + bz_);
      float r = sigf(xr + ar[v] + br_);
      float hh = tanhf(xh + r * (ah[v] + bh_));
      float hold = hF[m * 128 + n];
      float hn = z * hold + (1.0f - z) * hh;
      hF[m * 128 + n] = hn;
      hB[m * 136 + n] = (__bf16)hn;
      hs[((size_t)(b0 + m) * T + t) * ldh + coloff + n] = (__bf16)hn;
    }
    async_wait0();    // xp(t+1) landed
    __syncthreads();  // h update + xp(t+1) visible to all
  }
}

// ---------------------------------------------------------------------------
// Big GRU cell recurrence (units=1024). Grid: 4 blocks (16-row batch slices).
// U (6MB bf16) streamed from the 192MB L2 as prepacked fragments each step;
// h kept in LDS (fp32 master + bf16 WMMA copy + h_new double buffer).
// xp(t+1) async fetch overlapped with the h-commit phase. ~262KB LDS.
// ---------------------------------------------------------------------------
__global__ __launch_bounds__(256) void cell1024_kernel(
    const __bf16* __restrict__ xp,   // [B*T, 3072] bf16 (includes b[0])
    const __bf16* __restrict__ Upk,  // packed [192][32][32][16]
    const float* __restrict__ b1,    // [3072]
    __bf16* __restrict__ so,         // states out: row stride ldso, col offset
    int ldso, int coloff, int T) {
  extern __shared__ char smem[];
  float*  hF  = (float*)smem;                             // 16x1024 fp32 (64KB)
  __bf16* hB  = (__bf16*)(smem + 65536);                  // 16x1032 bf16 (33KB)
  float*  hN  = (float*)(smem + 65536 + 33024);           // 16x1024 fp32 (64KB)
  __bf16* xpl = (__bf16*)(smem + 65536 + 33024 + 65536);  // 16x3072 bf16 (96KB)

  const int tid = threadIdx.x, wave = tid >> 5, lane = tid & 31;
  const int b0 = blockIdx.x * 16;

  for (int i = tid; i < 16 * 1024; i += 256) hF[i] = 0.0f;
  for (int i = tid; i < 16 * 1032; i += 256) hB[i] = (__bf16)0.0f;
  for (int i = tid; i < 6144; i += 256) {  // async xp(0)
    int r = (i * 8) / 3072, c = (i * 8) % 3072;
    async_b128(xpl + r * 3072 + c, xp + ((size_t)(b0 + r) * T + 0) * 3072 + c);
  }
  async_wait0();
  __syncthreads();

  const int n16 = lane & 15;
  const int mb = (lane >> 4) * 8;

  for (int t = 0; t < T; ++t) {
#pragma unroll 1
    for (int j = 0; j < 8; ++j) {
      const int q = wave + 8 * j;  // z-column tile 0..63
      v8f az = {}, ar = {}, ah = {};
      for (int kt = 0; kt < 32; ++kt) {
        const int m = lane & 15;
        const int base = kt * 32 + ((lane >> 4) * 8);
        v8bf lo = *(const v8bf*)(hB + m * 1032 + base);
        v8bf hi = *(const v8bf*)(hB + m * 1032 + base + 16);
        v16bf a = __builtin_shufflevector(lo, hi, 0, 1, 2, 3, 4, 5, 6, 7, 8, 9,
                                          10, 11, 12, 13, 14, 15);
        const __bf16* pz = Upk + ((size_t)(q)       * 32 + kt) * 512 + lane * 16;
        const __bf16* pr = Upk + ((size_t)(q + 64)  * 32 + kt) * 512 + lane * 16;
        const __bf16* ph = Upk + ((size_t)(q + 128) * 32 + kt) * 512 + lane * 16;
        az = wmma_bf16(a, *(const v16bf*)pz, az);
        ar = wmma_bf16(a, *(const v16bf*)pr, ar);
        ah = wmma_bf16(a, *(const v16bf*)ph, ah);
      }
      const int n = q * 16 + n16;
      const float bz_ = b1[n], br_ = b1[1024 + n], bh_ = b1[2048 + n];
#pragma unroll
      for (int v = 0; v < 8; ++v) {
        const int m = mb + v;
        float xz = (float)xpl[m * 3072 + n];
        float xr = (float)xpl[m * 3072 + 1024 + n];
        float xh = (float)xpl[m * 3072 + 2048 + n];
        float z = sigf(xz + az[v] + bz_);
        float r = sigf(xr + ar[v] + br_);
        float hh = tanhf(xh + r * (ah[v] + bh_));
        float hold = hF[m * 1024 + n];
        hN[m * 1024 + n] = z * hold + (1.0f - z) * hh;  // deferred commit
      }
    }
    __syncthreads();  // xpl/hB/hF reads done; hN complete

    // prefetch xp(t+1) (async engine runs while VALU commits h)
    if (t + 1 < T) {
      for (int i = tid; i < 6144; i += 256) {
        int r = (i * 8) / 3072, c = (i * 8) % 3072;
        async_b128(xpl + r * 3072 + c,
                   xp + ((size_t)(b0 + r) * T + (t + 1)) * 3072 + c);
      }
    }
    // commit h_new -> h (fp32 + bf16) and emit states
    for (int i = tid; i < 16 * 1024; i += 256) {
      float hn = hN[i];
      int m = i >> 10, c = i & 1023;
      hF[i] = hn;
      hB[m * 1032 + c] = (__bf16)hn;
      so[((size_t)(b0 + m) * T + t) * ldso + coloff + c] = (__bf16)hn;
    }
    async_wait0();
    __syncthreads();
  }
}

// ---------------------------------------------------------------------------
// Host orchestration
// ---------------------------------------------------------------------------
static inline void launch_gemm(hipStream_t s, const __bf16* A, int lda,
                               const __bf16* Wp, int K, int N, const float* bias,
                               float* Cf, int ldcf, __bf16* Cb, int ldcb, int cboff,
                               __bf16* Cb2, int ldcb2, int cboff2, int act,
                               int M) {
  dim3 grid(N / 128, M / 64);
  gemm_bf16_kernel<<<grid, 256, 0, s>>>(A, lda, Wp, K / 32, bias, Cf, ldcf, Cb,
                                        ldcb, cboff, Cb2, ldcb2, cboff2, act);
}

static inline void launch_pack(hipStream_t s, const float* W, __bf16* out, int K,
                               int N) {
  int slots = (K * N) / 16;
  pack_w_kernel<<<(slots + 255) / 256, 256, 0, s>>>(W, out, K, N);
}

extern "C" void kernel_launch(void* const* d_in, const int* in_sizes, int n_in,
                              void* d_out, int out_size, void* d_ws, size_t ws_size,
                              hipStream_t stream) {
  (void)in_sizes; (void)n_in; (void)out_size; (void)ws_size;
  const int Tn = 512;
  const size_t M = (size_t)64 * 512;  // B*T = 32768

  const float* inputs = (const float*)d_in[0];
  const float* eps    = (const float*)d_in[1];
  const float* xrW  = (const float*)d_in[2];  const float* xrb  = (const float*)d_in[3];
  const float* g1W  = (const float*)d_in[4];  const float* g1U  = (const float*)d_in[5];
  const float* g1b  = (const float*)d_in[6];
  const float* g2W  = (const float*)d_in[7];  const float* g2U  = (const float*)d_in[8];
  const float* g2b  = (const float*)d_in[9];
  const float* xt1W = (const float*)d_in[10]; const float* xt1b = (const float*)d_in[11];
  const float* xt2W = (const float*)d_in[12]; const float* xt2b = (const float*)d_in[13];
  const float* zt1W = (const float*)d_in[14]; const float* zt1b = (const float*)d_in[15];
  const float* zt2W = (const float*)d_in[16]; const float* zt2b = (const float*)d_in[17];
  const float* ceW  = (const float*)d_in[18]; const float* ceU  = (const float*)d_in[19];
  const float* ceb  = (const float*)d_in[20];
  const float* p1W  = (const float*)d_in[21]; const float* p1b  = (const float*)d_in[22];
  const float* p2W  = (const float*)d_in[23]; const float* p2b  = (const float*)d_in[24];
  const float* pmW  = (const float*)d_in[25]; const float* pmb  = (const float*)d_in[26];
  const float* psW  = (const float*)d_in[27]; const float* psb  = (const float*)d_in[28];
  const float* t1W  = (const float*)d_in[29]; const float* t1b  = (const float*)d_in[30];
  const float* t2W  = (const float*)d_in[31]; const float* t2b  = (const float*)d_in[32];
  const float* tmW  = (const float*)d_in[33]; const float* tmb  = (const float*)d_in[34];
  const float* tsW  = (const float*)d_in[35]; const float* tsb  = (const float*)d_in[36];

  char* ws = (char*)d_ws;
  size_t off = 0;
  auto alloc = [&](size_t bytes) -> char* {
    char* p = ws + off;
    off = (off + bytes + 255) & ~(size_t)255;
    return p;
  };

  // packed weights (bf16)
  __bf16* pw_xr  = (__bf16*)alloc((size_t)256 * 128 * 2);
  __bf16* pw_g1W = (__bf16*)alloc((size_t)128 * 384 * 2);
  __bf16* pw_g1U = (__bf16*)alloc((size_t)128 * 384 * 2);
  __bf16* pw_g2W = (__bf16*)alloc((size_t)128 * 384 * 2);
  __bf16* pw_g2U = (__bf16*)alloc((size_t)128 * 384 * 2);
  __bf16* pw_xt1 = (__bf16*)alloc((size_t)128 * 512 * 2);
  __bf16* pw_xt2 = (__bf16*)alloc((size_t)512 * 512 * 2);
  __bf16* pw_zt1 = (__bf16*)alloc((size_t)128 * 512 * 2);
  __bf16* pw_zt2 = (__bf16*)alloc((size_t)512 * 512 * 2);
  __bf16* pw_ceW = (__bf16*)alloc((size_t)1024 * 3072 * 2);
  __bf16* pw_ceU = (__bf16*)alloc((size_t)1024 * 3072 * 2);
  __bf16* pw_p1  = (__bf16*)alloc((size_t)512 * 512 * 2);  // only rows 0..511 (h0==0)
  __bf16* pw_p2  = (__bf16*)alloc((size_t)512 * 512 * 2);
  __bf16* pw_pm  = (__bf16*)alloc((size_t)512 * 128 * 2);
  __bf16* pw_ps  = (__bf16*)alloc((size_t)512 * 128 * 2);
  __bf16* pw_t1  = (__bf16*)alloc((size_t)1536 * 512 * 2);
  __bf16* pw_t2  = (__bf16*)alloc((size_t)512 * 512 * 2);
  __bf16* pw_tm  = (__bf16*)alloc((size_t)512 * 128 * 2);
  __bf16* pw_ts  = (__bf16*)alloc((size_t)512 * 128 * 2);

  // activations
  __bf16* in_bf = (__bf16*)alloc(M * 256 * 2);
  __bf16* bufA  = (__bf16*)alloc(M * 128 * 2);   // X1 -> hs2
  __bf16* bufB  = (__bf16*)alloc(M * 128 * 2);   // hs1 -> z
  float*  xpS   = (float*)alloc(M * 384 * 4);    // gru1 xp then gru2 xp
  __bf16* bufE  = (__bf16*)alloc(M * 512 * 2);   // X2 -> zprime_mid
  __bf16* cxz   = (__bf16*)alloc(M * 1024 * 2);  // concat([X, zprime])
  __bf16* bufG  = (__bf16*)alloc(M * 512 * 2);
  __bf16* bufH  = (__bf16*)alloc(M * 512 * 2);
  float*  muF   = (float*)alloc(M * 128 * 4);
  float*  sigF  = (float*)alloc(M * 128 * 4);
  __bf16* czs   = (__bf16*)alloc(M * 1536 * 2);  // concat([zprime, states])
  __bf16* cxp   = (__bf16*)alloc(M * 3072 * 2);  // cell input projection

  float* outMu  = (float*)d_out;
  float* outSig = (float*)d_out + M * 128;

  // --- weight prepack ---
  launch_pack(stream, xrW,  pw_xr,  256, 128);
  launch_pack(stream, g1W,  pw_g1W, 128, 384);
  launch_pack(stream, g1U,  pw_g1U, 128, 384);
  launch_pack(stream, g2W,  pw_g2W, 128, 384);
  launch_pack(stream, g2U,  pw_g2U, 128, 384);
  launch_pack(stream, xt1W, pw_xt1, 128, 512);
  launch_pack(stream, xt2W, pw_xt2, 512, 512);
  launch_pack(stream, zt1W, pw_zt1, 128, 512);
  launch_pack(stream, zt2W, pw_zt2, 512, 512);
  launch_pack(stream, ceW,  pw_ceW, 1024, 3072);
  launch_pack(stream, ceU,  pw_ceU, 1024, 3072);
  launch_pack(stream, p1W,  pw_p1,  512, 512);
  launch_pack(stream, p2W,  pw_p2,  512, 512);
  launch_pack(stream, pmW,  pw_pm,  512, 128);
  launch_pack(stream, psW,  pw_ps,  512, 128);
  launch_pack(stream, t1W,  pw_t1,  1536, 512);
  launch_pack(stream, t2W,  pw_t2,  512, 512);
  launch_pack(stream, tmW,  pw_tm,  512, 128);
  launch_pack(stream, tsW,  pw_ts,  512, 128);

  // --- inputs -> bf16 ---
  {
    size_t n = M * 256;
    f2bf_kernel<<<(unsigned)((n + 255) / 256), 256, 0, stream>>>(inputs, in_bf, n);
  }

  const size_t smem_gru  = 98304 + 8192 + 4352 + 2 * 24576;  // 160,000 B
  const size_t smem_cell = 65536 + 33024 + 65536 + 98304;    // 262,400 B

  // X = relu(inputs @ xr_W + b)
  launch_gemm(stream, in_bf, 256, pw_xr, 256, 128, xrb,
              nullptr, 0, bufA, 128, 0, nullptr, 0, 0, 1, (int)M);
  // gru1: xp then recurrence
  launch_gemm(stream, bufA, 128, pw_g1W, 128, 384, g1b,
              xpS, 384, nullptr, 0, 0, nullptr, 0, 0, 0, (int)M);
  gru128_kernel<<<4, 256, smem_gru, stream>>>(xpS, pw_g1U, g1b + 384,
                                              bufB, 128, 0, Tn);
  // gru2
  launch_gemm(stream, bufB, 128, pw_g2W, 128, 384, g2b,
              xpS, 384, nullptr, 0, 0, nullptr, 0, 0, 0, (int)M);
  gru128_kernel<<<4, 256, smem_gru, stream>>>(xpS, pw_g2U, g2b + 384,
                                              bufA, 128, 0, Tn);
  // X_transform
  launch_gemm(stream, bufA, 128, pw_xt1, 128, 512, xt1b,
              nullptr, 0, bufE, 512, 0, nullptr, 0, 0, 1, (int)M);
  launch_gemm(stream, bufE, 512, pw_xt2, 512, 512, xt2b,
              nullptr, 0, cxz, 1024, 0, nullptr, 0, 0, 1, (int)M);
  // phi (conditions on zero h0: K=512 only)
  launch_gemm(stream, cxz, 1024, pw_p1, 512, 512, p1b,
              nullptr, 0, bufG, 512, 0, nullptr, 0, 0, 1, (int)M);
  launch_gemm(stream, bufG, 512, pw_p2, 512, 512, p2b,
              nullptr, 0, bufH, 512, 0, nullptr, 0, 0, 1, (int)M);
  launch_gemm(stream, bufH, 512, pw_pm, 512, 128, pmb,
              muF, 128, nullptr, 0, 0, nullptr, 0, 0, 0, (int)M);
  launch_gemm(stream, bufH, 512, pw_ps, 512, 128, psb,
              sigF, 128, nullptr, 0, 0, nullptr, 0, 0, 2, (int)M);
  {
    size_t n = M * 128;
    z_kernel<<<(unsigned)((n + 255) / 256), 256, 0, stream>>>(muF, sigF, eps, bufB, n);
  }
  // Z_transform; zprime goes to both concat buffers
  launch_gemm(stream, bufB, 128, pw_zt1, 128, 512, zt1b,
              nullptr, 0, bufE, 512, 0, nullptr, 0, 0, 1, (int)M);
  launch_gemm(stream, bufE, 512, pw_zt2, 512, 512, zt2b,
              nullptr, 0, czs, 1536, 0, cxz, 1024, 512, 1, (int)M);
  // cell: input projection (time-parallel), then sequential recurrence
  launch_gemm(stream, cxz, 1024, pw_ceW, 1024, 3072, ceb,
              nullptr, 0, cxp, 3072, 0, nullptr, 0, 0, 0, (int)M);
  cell1024_kernel<<<4, 256, smem_cell, stream>>>(cxp, pw_ceU, ceb + 3072,
                                                 czs, 1536, 512, Tn);
  // theta
  launch_gemm(stream, czs, 1536, pw_t1, 1536, 512, t1b,
              nullptr, 0, bufG, 512, 0, nullptr, 0, 0, 1, (int)M);
  launch_gemm(stream, bufG, 512, pw_t2, 512, 512, t2b,
              nullptr, 0, bufH, 512, 0, nullptr, 0, 0, 1, (int)M);
  launch_gemm(stream, bufH, 512, pw_tm, 512, 128, tmb,
              outMu, 128, nullptr, 0, 0, nullptr, 0, 0, 0, (int)M);
  launch_gemm(stream, bufH, 512, pw_ts, 512, 128, tsb,
              outSig, 128, nullptr, 0, 0, nullptr, 0, 0, 2, (int)M);
}
